// GALanguageModel_13383118094732
// MI455X (gfx1250) — compile-verified, matrix operand order checked by reading, not run on hardware
//
#include <hip/hip_runtime.h>
#include <hip/hip_bf16.h>

// ---------------------------------------------------------------------------
// GA language model forward for gfx1250 (MI455X).  All GEMM-shaped work runs
// through v_wmma_f32_16x16x32_f16 (wave32; each wave computes a 16x64 output
// strip = 4 accumulators, reusing the A fragment across 4 B fragments).
// ---------------------------------------------------------------------------

typedef __attribute__((ext_vector_type(16))) _Float16 v16h;
typedef __attribute__((ext_vector_type(8)))  float    v8f;

#define NTOK 2048      // B*L tokens
#define SEQL 1024
#define NBATCH 2
#define NC   64        // channels
#define ND   16        // blades
#define NH   8         // heads
#define NCH  8         // channels per head
#define HD   128       // NCH*ND
#define CD   1024      // NC*ND
#define NCE  256       // expanded channels
#define NV   32000
#define NKR  8         // rotors
#define NL   2

enum {
  MODE_Q = 0, MODE_K = 1, MODE_V = 2, MODE_SCORES = 3,
  MODE_ATTNOUT = 4, MODE_ADDX = 5, MODE_H = 6, MODE_LOGITS = 7
};

// Cayley sign for Cl(3,1): sign(a,b) with metric (+,+,+,-)
__device__ __forceinline__ float cay_sign(int a, int b) {
  int tot = 0;
  for (int t = a >> 1; t; t >>= 1) tot += __popc(t & b);
  if ((a & b) & 8) tot++;               // e3*e3 = -1 (bit 3)
  return (tot & 1) ? -1.0f : 1.0f;
}

__device__ __forceinline__ float rev_sign(int d) {
  int g = __popc(d);
  return ((g * (g - 1) / 2) & 1) ? -1.0f : 1.0f;
}

__device__ __forceinline__ float gelu_f(float v) {
  float v3 = v * v * v;
  return 0.5f * v * (1.0f + tanhf(0.7978845608f * (v + 0.044715f * v3)));
}

// ---------------------------------------------------------------------------
// f32 -> f16 weight conversion
// ---------------------------------------------------------------------------
__global__ void f32_to_f16_kernel(const float* __restrict__ s,
                                  _Float16* __restrict__ d, int n) {
  int i = blockIdx.x * 256 + threadIdx.x;
  if (i < n) d[i] = (_Float16)s[i];
}

// ---------------------------------------------------------------------------
// Embedding gather + rotary bivector PE (rotor has only blades {0,3}):
//   y[b] = c0*x[b] + s3*sign(3,3^b)*x[3^b];  z[b] = c0*y[b] - s3*sign(b^3,3)*y[b^3]
// ---------------------------------------------------------------------------
__global__ void embed_rotary_kernel(const int* __restrict__ tok,
                                    const float* __restrict__ embed,
                                    float* __restrict__ x) {
  int idx = blockIdx.x * 256 + threadIdx.x;          // one (token, channel)
  if (idx >= NTOK * NC) return;
  int n = idx >> 6, c = idx & 63;
  int l = n & (SEQL - 1);
  int tid = tok[n];
  const float* e = embed + ((size_t)tid * NC + c) * ND;
  float v[16], y[16];
#pragma unroll
  for (int d = 0; d < 16; ++d) v[d] = e[d];
  float freq = __expf(-9.210340372f * (float)c * (1.0f / 64.0f)); // 10000^(-c/64)
  float th = (float)l * freq;
  float c0 = cosf(th), s3 = sinf(th);
#pragma unroll
  for (int d = 0; d < 16; ++d)
    y[d] = c0 * v[d] + s3 * cay_sign(3, 3 ^ d) * v[3 ^ d];
  float* xp = x + (size_t)n * CD + c * ND;
#pragma unroll
  for (int d = 0; d < 16; ++d)
    xp[d] = c0 * y[d] - s3 * cay_sign(d ^ 3, 3) * y[d ^ 3];
}

// ---------------------------------------------------------------------------
// CliffordLayerNorm -> transposed f16 output Xc[(n*16+d)*64 + c]
// (B-operand layout for the channel-mixing WMMA GEMMs)
// ---------------------------------------------------------------------------
__global__ void cln_colmajor_kernel(const float* __restrict__ x,
                                    const float* __restrict__ g,
                                    _Float16* __restrict__ Xc) {
  int n = blockIdx.x, t = threadIdx.x;
  __shared__ float red[256];
  const float* xp = x + (size_t)n * CD;
  float ss = 0.f;
  for (int i = t; i < CD; i += 256) { float v = xp[i]; ss += v * v; }
  red[t] = ss; __syncthreads();
  for (int s = 128; s > 0; s >>= 1) { if (t < s) red[t] += red[t + s]; __syncthreads(); }
  float inv = rsqrtf(red[0] * (1.0f / (float)CD) + 1e-6f);
  for (int i = t; i < CD; i += 256) {
    int c = i >> 4, d = i & 15;
    Xc[(((size_t)n * ND + d) << 6) + c] = (_Float16)(xp[i] * inv * g[c]);
  }
}

// ---------------------------------------------------------------------------
// Final LN + blade-0 sigmoid gate -> f16 A-operand g0[n*64 + c] for vocab head
// ---------------------------------------------------------------------------
__global__ void head_prep_kernel(const float* __restrict__ x,
                                 const float* __restrict__ g,
                                 const float* __restrict__ sel,
                                 _Float16* __restrict__ g0) {
  int n = blockIdx.x, t = threadIdx.x;
  __shared__ float red[256];
  const float* xp = x + (size_t)n * CD;
  float ss = 0.f;
  for (int i = t; i < CD; i += 256) { float v = xp[i]; ss += v * v; }
  red[t] = ss; __syncthreads();
  for (int s = 128; s > 0; s >>= 1) { if (t < s) red[t] += red[t + s]; __syncthreads(); }
  float inv = rsqrtf(red[0] * (1.0f / (float)CD) + 1e-6f);
  if (t < NC) {
    float gate = 1.0f / (1.0f + __expf(-sel[t * ND + 0]));
    g0[(size_t)n * NC + t] = (_Float16)(xp[t * ND + 0] * inv * g[t] * gate);
  }
}

// ---------------------------------------------------------------------------
// Precompute per-layer 16x16 mix matrix Tmix = sum_k softmax(mixw)_k * M_L(R_k)M_R(R~_k)
// ---------------------------------------------------------------------------
__global__ void rotor_prep_kernel(const float* __restrict__ rotor_biv,
                                  const float* __restrict__ mixw,
                                  float* __restrict__ Tmix) {
  int l = blockIdx.x, t = threadIdx.x;
  __shared__ float rot[NKR][16];
  __shared__ float w[NKR];
  if (t < NKR) {
    const int BIV[6] = {3, 5, 6, 9, 10, 12};
    float r[16];
#pragma unroll
    for (int d = 0; d < 16; ++d) r[d] = 0.f;
    r[0] = 1.f;
    for (int i = 0; i < 6; ++i) r[BIV[i]] = rotor_biv[(l * NKR + t) * 6 + i];
    float nn = 0.f;
#pragma unroll
    for (int d = 0; d < 16; ++d) nn += r[d] * r[d];
    float inv = rsqrtf(nn);
#pragma unroll
    for (int d = 0; d < 16; ++d) rot[t][d] = r[d] * inv;
  }
  if (t == 0) {
    float mx = -3.4e38f;
    for (int k = 0; k < NKR; ++k) mx = fmaxf(mx, mixw[l * NKR + k]);
    float sm = 0.f;
    for (int k = 0; k < NKR; ++k) { w[k] = __expf(mixw[l * NKR + k] - mx); sm += w[k]; }
    for (int k = 0; k < NKR; ++k) w[k] /= sm;
  }
  __syncthreads();
  // thread t -> (m, j) entry of Tmix
  int m = t >> 4, j = t & 15;
  const int COMPS[7] = {0, 3, 5, 6, 9, 10, 12};
  const int CMASK = 0x1669;  // bits 0,3,5,6,9,10,12
  float acc = 0.f;
  for (int k = 0; k < NKR; ++k) {
#pragma unroll
    for (int ai = 0; ai < 7; ++ai) {
      int a = COMPS[ai];
      int b = a ^ j ^ m;
      if (!((CMASK >> b) & 1)) continue;
      float s = cay_sign(a, j) * cay_sign(a ^ j, b);
      acc += w[k] * rot[k][a] * (rot[k][b] * rev_sign(b)) * s;
    }
  }
  Tmix[l * 256 + t] = acc;
}

// ---------------------------------------------------------------------------
// FFN middle: per (n,e) apply Tmix (16x16) then gelu, write transposed f16
// Hc[(n*16+d)*256 + e] (B-operand for ffn_down GEMM)
// ---------------------------------------------------------------------------
__global__ void ffn_mid_kernel(const float* __restrict__ hbuf,
                               const float* __restrict__ Tmix,
                               _Float16* __restrict__ Hc) {
  __shared__ float T[256];
  T[threadIdx.x] = Tmix[threadIdx.x];
  __syncthreads();
  int idx = blockIdx.x * 256 + threadIdx.x;          // (n, e)
  if (idx >= NTOK * NCE) return;
  int n = idx >> 8, e = idx & 255;
  const float* hp = hbuf + (size_t)idx * ND;
  float hin[16];
#pragma unroll
  for (int d = 0; d < 16; ++d) hin[d] = hp[d];
#pragma unroll
  for (int d = 0; d < 16; ++d) {
    float acc = 0.f;
#pragma unroll
    for (int j = 0; j < 16; ++j) acc += T[d * 16 + j] * hin[j];
    Hc[((size_t)n * ND + d) * NCE + e] = (_Float16)gelu_f(acc);
  }
}

// ---------------------------------------------------------------------------
// Row-wise causal softmax; writes f16 attn row (A-operand of AV GEMM), zeros
// in the masked region.
// ---------------------------------------------------------------------------
__global__ void softmax_kernel(const float* __restrict__ S,
                               _Float16* __restrict__ A) {
  int r = blockIdx.x;                 // r = bh*SEQL + l
  int l = r & (SEQL - 1);
  const float* s = S + (size_t)r * SEQL;
  _Float16* a = A + (size_t)r * SEQL;
  int t = threadIdx.x;
  __shared__ float red[256];
  float mx = -3.4e38f;
  for (int m = t; m <= l; m += 256) mx = fmaxf(mx, s[m]);
  red[t] = mx; __syncthreads();
  for (int st = 128; st > 0; st >>= 1) { if (t < st) red[t] = fmaxf(red[t], red[t + st]); __syncthreads(); }
  mx = red[0]; __syncthreads();
  float sum = 0.f;
  for (int m = t; m <= l; m += 256) sum += __expf(s[m] - mx);
  red[t] = sum; __syncthreads();
  for (int st = 128; st > 0; st >>= 1) { if (t < st) red[t] += red[t + st]; __syncthreads(); }
  float inv = 1.0f / red[0];
  for (int m = t; m < SEQL; m += 256) {
    float v = (m <= l) ? __expf(s[m] - mx) * inv : 0.f;
    a[m] = (_Float16)v;
  }
}

// ---------------------------------------------------------------------------
// Generic WMMA GEMM: each wave computes a 16x64 output strip (4 accumulators,
// A fragment reused across the 4 N sub-tiles), K in steps of 32 (f16).
//   A: f16, element (m,k) at A[m*lda + k]            (row-major)
//   B: f16, element (k,j) at B[j*ldb + k]            ("col-major" [j][k])
// Fragment packing follows the documented 16-bit A layout (lane&15 selects
// the M/N index, lane>>4 selects the K half, packed (k,k+1) dword pairs).
// Output routing selected by `mode`.  Requires M%16==0, N%64==0, K%32==0.
// ---------------------------------------------------------------------------
__global__ void wmma_gemm_kernel(const _Float16* __restrict__ A, long long sA, int lda,
                                 const _Float16* __restrict__ B, long long sB, int ldb,
                                 void* __restrict__ Y, const float* __restrict__ aux,
                                 int M, int N, int Kdim, float alpha, int mode) {
  const int lane = threadIdx.x & 31;
  const int wave = threadIdx.x >> 5;
  const int lane15 = lane & 15;
  const int tilesM = M >> 4, strpN = N >> 6;
  long long tileId = (long long)blockIdx.x * 8 + wave;
  if (tileId >= (long long)tilesM * strpN) return;
  int tm = (int)(tileId % tilesM);
  int tq = (int)(tileId / tilesM);
  int rowBase = tm << 4, colBase = tq << 6;

  const _Float16* Ap = A + (long long)blockIdx.y * sA + (size_t)(rowBase + lane15) * lda;
  const _Float16* Bbase = B + (long long)blockIdx.y * sB;
  const _Float16* Bp[4];
#pragma unroll
  for (int i = 0; i < 4; ++i)
    Bp[i] = Bbase + (size_t)(colBase + 16 * i + lane15) * ldb;
  const int khalf = (lane >> 4) << 3;   // 0 or 8

  v8f acc[4];
#pragma unroll
  for (int i = 0; i < 4; ++i) acc[i] = (v8f){0.f, 0.f, 0.f, 0.f, 0.f, 0.f, 0.f, 0.f};
  union Frag { v16h v; unsigned u[8]; };

  for (int kt = 0; kt < Kdim; kt += 32) {
    Frag a, b[4];
#pragma unroll
    for (int vv = 0; vv < 8; ++vv) {
      int k = kt + ((vv < 4) ? (2 * vv) : (8 + 2 * vv)) + khalf;
      a.u[vv] = *(const unsigned*)(Ap + k);
#pragma unroll
      for (int i = 0; i < 4; ++i)
        b[i].u[vv] = *(const unsigned*)(Bp[i] + k);
    }
    if (kt + 32 < Kdim) {               // uniform branch: prefetch next k-step
      __builtin_prefetch(Ap + kt + 32 + khalf, 0, 1);
#pragma unroll
      for (int i = 0; i < 4; ++i)
        __builtin_prefetch(Bp[i] + kt + 32 + khalf, 0, 1);
    }
#pragma unroll
    for (int i = 0; i < 4; ++i)
      acc[i] = __builtin_amdgcn_wmma_f32_16x16x32_f16(false, a.v, false, b[i].v,
                                                      (short)0, acc[i], false, false);
  }

  const int rhalf = (lane >> 4) ? 8 : 0;
#pragma unroll
  for (int i = 0; i < 4; ++i) {
    const int gcol = colBase + 16 * i + lane15;
#pragma unroll
    for (int r = 0; r < 8; ++r) {
      int grow = rowBase + r + rhalf;
      float val = acc[i][r] * alpha;
      switch (mode) {
        case MODE_Q:
        case MODE_K: {
          int n = gcol >> 4, d = gcol & 15;
          int hh = grow >> 3, c8 = grow & 7;
          int bb = n >> 10, l = n & (SEQL - 1);
          if (mode == MODE_K) val *= cay_sign(d, d);     // fold GMET signs
          ((_Float16*)Y)[(((size_t)(bb * NH + hh) * SEQL + l) * HD) + c8 * ND + d] = (_Float16)val;
          break;
        }
        case MODE_V: {
          int n = gcol >> 4, d = gcol & 15;
          int hh = grow >> 3, c8 = grow & 7;
          int bb = n >> 10, l = n & (SEQL - 1);
          ((_Float16*)Y)[(((size_t)(bb * NH + hh) * HD) + c8 * ND + d) * SEQL + l] = (_Float16)val;
          break;
        }
        case MODE_SCORES: {
          ((float*)Y)[((size_t)blockIdx.y * SEQL + grow) * SEQL + gcol] = val;
          break;
        }
        case MODE_ATTNOUT: {
          int bb = blockIdx.y >> 3, hh = blockIdx.y & 7;
          int n = bb * SEQL + grow;
          int c8 = gcol >> 4, d = gcol & 15;
          ((_Float16*)Y)[((size_t)n * ND + d) * NC + hh * NCH + c8] = (_Float16)val;
          break;
        }
        case MODE_ADDX: {
          int n = gcol >> 4, d = gcol & 15;
          float* p = (float*)Y + (size_t)n * CD + grow * ND + d;
          *p += val;                                     // residual add
          break;
        }
        case MODE_H: {
          int n = gcol >> 4, d = gcol & 15;
          ((float*)Y)[(size_t)n * (NCE * ND) + grow * ND + d] = val;
          break;
        }
        case MODE_LOGITS: {
          ((float*)Y)[(size_t)grow * NV + gcol] = val + aux[gcol];
          break;
        }
      }
    }
  }
}

// ---------------------------------------------------------------------------
// Host orchestration
// ---------------------------------------------------------------------------
extern "C" void kernel_launch(void* const* d_in, const int* in_sizes, int n_in,
                              void* d_out, int out_size, void* d_ws, size_t ws_size,
                              hipStream_t stream) {
  const int*   tok      = (const int*)  d_in[0];
  const float* embed    = (const float*)d_in[1];
  const float* ln1_g    = (const float*)d_in[2];
  const float* ln2_g    = (const float*)d_in[3];
  const float* wq       = (const float*)d_in[4];
  const float* wk       = (const float*)d_in[5];
  const float* wv       = (const float*)d_in[6];
  const float* wo       = (const float*)d_in[7];
  const float* ffn_up   = (const float*)d_in[8];
  const float* ffn_down = (const float*)d_in[9];
  const float* rotorb   = (const float*)d_in[10];
  const float* mixw     = (const float*)d_in[11];
  const float* out_ln_g = (const float*)d_in[12];
  const float* sel      = (const float*)d_in[13];
  const float* head_w   = (const float*)d_in[14];
  const float* head_b   = (const float*)d_in[15];
  float* out = (float*)d_out;

  char* ws = (char*)d_ws;
  size_t off = 0;
  auto alloc = [&](size_t bytes) -> void* {
    void* p = ws + off;
    off += (bytes + 255) & ~(size_t)255;
    return p;
  };

  float*    x      = (float*)   alloc((size_t)NTOK * CD * 4);         // 8 MB
  _Float16* Xc     = (_Float16*)alloc((size_t)NTOK * CD * 2);         // 4 MB (reused ln1/ln2)
  _Float16* qf     = (_Float16*)alloc((size_t)NBATCH * NH * SEQL * HD * 2);
  _Float16* kf     = (_Float16*)alloc((size_t)NBATCH * NH * SEQL * HD * 2);
  _Float16* vt     = (_Float16*)alloc((size_t)NBATCH * NH * HD * SEQL * 2);
  float*    scores = (float*)   alloc((size_t)NBATCH * NH * SEQL * SEQL * 4);  // 64 MB
  _Float16* attn   = (_Float16*)alloc((size_t)NBATCH * NH * SEQL * SEQL * 2);  // 32 MB
  _Float16* Oc     = (_Float16*)alloc((size_t)NTOK * CD * 2);
  float*    hbuf   = (float*)   alloc((size_t)NTOK * NCE * ND * 4);   // 32 MB
  _Float16* Hc     = (_Float16*)alloc((size_t)NTOK * NCE * ND * 2);   // 16 MB
  _Float16* g0     = (_Float16*)alloc((size_t)NTOK * NC * 2);
  _Float16* wq_h   = (_Float16*)alloc((size_t)NL * NC * NC * 2);
  _Float16* wk_h   = (_Float16*)alloc((size_t)NL * NC * NC * 2);
  _Float16* wv_h   = (_Float16*)alloc((size_t)NL * NC * NC * 2);
  _Float16* wo_h   = (_Float16*)alloc((size_t)NL * NC * NC * 2);
  _Float16* up_h   = (_Float16*)alloc((size_t)NL * NCE * NC * 2);
  _Float16* dn_h   = (_Float16*)alloc((size_t)NL * NC * NCE * 2);
  _Float16* hw_h   = (_Float16*)alloc((size_t)NV * NC * 2);
  float*    Tmix   = (float*)   alloc((size_t)NL * 256 * 4);
  if (off > ws_size) return;   // workspace too small: bail deterministically

  auto conv = [&](const float* s, _Float16* d, int n) {
    f32_to_f16_kernel<<<(n + 255) / 256, 256, 0, stream>>>(s, d, n);
  };
  auto gemm = [&](const _Float16* A, long long sA, int lda,
                  const _Float16* B, long long sB, int ldb,
                  void* Y, const float* aux, int M, int N, int K,
                  float alpha, int mode, int batches) {
    long long tiles = (long long)(M / 16) * (N / 64);   // 16x64 strips per wave
    dim3 grid((unsigned)((tiles + 7) / 8), batches, 1);
    wmma_gemm_kernel<<<grid, 256, 0, stream>>>(A, sA, lda, B, sB, ldb, Y, aux,
                                               M, N, K, alpha, mode);
  };

  // weight conversions + rotor mix matrices
  conv(wq, wq_h, NL * NC * NC);
  conv(wk, wk_h, NL * NC * NC);
  conv(wv, wv_h, NL * NC * NC);
  conv(wo, wo_h, NL * NC * NC);
  conv(ffn_up, up_h, NL * NCE * NC);
  conv(ffn_down, dn_h, NL * NC * NCE);
  conv(head_w, hw_h, NV * NC);
  rotor_prep_kernel<<<NL, 256, 0, stream>>>(rotorb, mixw, Tmix);

  // embedding + rotary PE
  embed_rotary_kernel<<<(NTOK * NC + 255) / 256, 256, 0, stream>>>(tok, embed, x);

  const int NCOL = NTOK * ND;             // 32768 columns for channel linears
  const float scale = 0.08838834764831845f;  // 1/sqrt(ch*D) = 1/sqrt(128)

  for (int l = 0; l < NL; ++l) {
    // --- attention ---
    cln_colmajor_kernel<<<NTOK, 256, 0, stream>>>(x, ln1_g + l * NC, Xc);
    gemm(wq_h + l * NC * NC, 0, NC, Xc, 0, NC, qf, nullptr, NC, NCOL, NC, 1.f, MODE_Q, 1);
    gemm(wk_h + l * NC * NC, 0, NC, Xc, 0, NC, kf, nullptr, NC, NCOL, NC, 1.f, MODE_K, 1);
    gemm(wv_h + l * NC * NC, 0, NC, Xc, 0, NC, vt, nullptr, NC, NCOL, NC, 1.f, MODE_V, 1);
    // scores[bh] = Q[bh] (L x 128) * K[bh]^T  (GMET folded into kf)
    gemm(qf, (long long)SEQL * HD, HD, kf, (long long)SEQL * HD, HD,
         scores, nullptr, SEQL, SEQL, HD, scale, MODE_SCORES, NBATCH * NH);
    softmax_kernel<<<NBATCH * NH * SEQL, 256, 0, stream>>>(scores, attn);
    // O[bh] = attn (L x L) * V[bh] (L x 128)
    gemm(attn, (long long)SEQL * SEQL, SEQL, vt, (long long)HD * SEQL, SEQL,
         Oc, nullptr, SEQL, HD, SEQL, 1.f, MODE_ATTNOUT, NBATCH * NH);
    // x += wo @ O
    gemm(wo_h + l * NC * NC, 0, NC, Oc, 0, NC, x, nullptr, NC, NCOL, NC, 1.f, MODE_ADDX, 1);

    // --- FFN ---
    cln_colmajor_kernel<<<NTOK, 256, 0, stream>>>(x, ln2_g + l * NC, Xc);
    gemm(up_h + l * NCE * NC, 0, NC, Xc, 0, NC, hbuf, nullptr, NCE, NCOL, NC, 1.f, MODE_H, 1);
    ffn_mid_kernel<<<(NTOK * NCE + 255) / 256, 256, 0, stream>>>(hbuf, Tmix + l * 256, Hc);
    gemm(dn_h + l * NC * NCE, 0, NCE, Hc, 0, NCE, x, nullptr, NC, NCOL, NCE, 1.f, MODE_ADDX, 1);
  }

  // final norm + blade gate + vocab head
  head_prep_kernel<<<NTOK, 256, 0, stream>>>(x, out_ln_g, sel, g0);
  gemm(g0, 0, NC, hw_h, 0, NC, out, head_b, NTOK, NV, NC, 1.f, MODE_LOGITS, 1);
}
